// VectorQuantize_25589415149648
// MI455X (gfx1250) — matrix operand order repaired and checked by reading, not compiled, not run on hardware
//
#include <hip/hip_runtime.h>
#include <hip/hip_bf16.h>

// ---------------------------------------------------------------------------
// VQ: dist = 2 x.e - |x|^2 - |e|^2 ; argmax over K=2048 ; gather fp32 codeword
// N = 32768 rows, d = 512, K = 2048.
// GEMM runs in bf16 via v_wmma_f32_16x16x32_bf16 (fp32 accumulate); outputs /
// loss use the fp32 tables. Outputs concatenated: [quantize 16777216][hist
// 2048][loss 1][perplexity 1].
// ---------------------------------------------------------------------------

typedef __attribute__((ext_vector_type(16))) __bf16 v16bf;
typedef __attribute__((ext_vector_type(8)))  float  v8f;

#define VQ_D        512
#define VQ_K        2048
#define VQ_N        32768
#define VQ_KCHUNKS  16      // 512 / 32
#define VQ_NTILES   128     // 2048 / 16
#define LDS_STRIDE  260     // 256 uints (512 bf16) + 4 pad -> bank-conflict free

static __device__ __forceinline__ unsigned short bf_bits(float f) {
  union { float f; unsigned u; } a; a.f = f;
  unsigned r = a.u + 0x7fffu + ((a.u >> 16) & 1u);   // round-to-nearest-even
  return (unsigned short)(r >> 16);
}

__global__ void vq_init_kernel(float* stats) {
  int i = threadIdx.x + blockIdx.x * blockDim.x;
  if (i < VQ_K + 1) stats[i] = 0.0f;   // hist[2048] + raw loss slot
}

__global__ __launch_bounds__(256)
void vq_main_kernel(const float* __restrict__ x, const float* __restrict__ embed,
                    float* __restrict__ outq, float* __restrict__ hist,
                    float* __restrict__ lossSlot) {
  __shared__ unsigned sB[16 * LDS_STRIDE];  // bf16 embed tile, packed pairs
  __shared__ float    sE2[16];              // |e|^2 per code in tile
  __shared__ int      sIdx[8 * 16];         // best code per row, per wave

  const int tid    = threadIdx.x;
  const int lane   = tid & 31;
  const int wave   = tid >> 5;
  const int m      = lane & 15;               // row (A) / code col (B) in tile
  const int kbaseA = (lane < 16) ? 0 : 8;     // A-fragment K sub-lane base
  const int kbaseB = (lane < 16) ? 0 : 8;     // B-fragment K sub-lane base
  const int rowBase = blockIdx.x * 128 + wave * 16;

  // ---- Build 16 bf16 A fragments (rows rowBase..rowBase+15), kept in VGPRs.
  // 16-bit A 16x32 layout: lanes 0-15 hold K={0..7,16..23}, lanes 16-31 hold
  // K={8..15,24..31} of the 32-wide chunk, two contiguous 8-float runs each.
  union AU { unsigned short s[16]; v16bf v; };
  AU afrag[VQ_KCHUNKS];
  {
    const float* xrow = x + (size_t)(rowBase + m) * VQ_D;
#pragma unroll
    for (int kc = 0; kc < VQ_KCHUNKS; ++kc) {
      const float* p0 = xrow + kc * 32 + kbaseA;
      float4 a0 = ((const float4*)p0)[0];
      float4 a1 = ((const float4*)p0)[1];
      float4 b0 = ((const float4*)(p0 + 16))[0];
      float4 b1 = ((const float4*)(p0 + 16))[1];
      float v[16] = {a0.x,a0.y,a0.z,a0.w, a1.x,a1.y,a1.z,a1.w,
                     b0.x,b0.y,b0.z,b0.w, b1.x,b1.y,b1.z,b1.w};
#pragma unroll
      for (int i = 0; i < 16; ++i) afrag[kc].s[i] = bf_bits(v[i]);
    }
  }

  float best[8];
  int   bidx[8];
#pragma unroll
  for (int j = 0; j < 8; ++j) { best[j] = -3.4e38f; bidx[j] = 0; }

  const int r   = tid >> 4;   // cooperative-load row     0..15
  const int seg = tid & 15;   // cooperative-load segment 0..15 (32 floats each)

  for (int tile = 0; tile < VQ_NTILES; ++tile) {
    // ---- Cooperative: fp32 embed tile -> bf16 in LDS, plus |e|^2 per code.
    {
      const float* ep = embed + ((size_t)(tile * 16 + r)) * VQ_D + seg * 32;
      float sumsq = 0.0f;
      unsigned pk[16];
#pragma unroll
      for (int q = 0; q < 8; ++q) {
        float4 f = ((const float4*)ep)[q];
        sumsq += f.x*f.x + f.y*f.y + f.z*f.z + f.w*f.w;
        pk[q*2+0] = (unsigned)bf_bits(f.x) | ((unsigned)bf_bits(f.y) << 16);
        pk[q*2+1] = (unsigned)bf_bits(f.z) | ((unsigned)bf_bits(f.w) << 16);
      }
      unsigned* dst = &sB[r * LDS_STRIDE + seg * 16];
#pragma unroll
      for (int q = 0; q < 4; ++q)
        ((uint4*)dst)[q] = make_uint4(pk[q*4+0], pk[q*4+1], pk[q*4+2], pk[q*4+3]);
#pragma unroll
      for (int mask = 1; mask <= 8; mask <<= 1)
        sumsq += __shfl_xor(sumsq, mask, 32);
      if (seg == 0) sE2[r] = sumsq;
      if (tile + 1 < VQ_NTILES)   // global_prefetch_b8 for next tile
        __builtin_prefetch(embed + ((size_t)((tile + 1) * 16 + r)) * VQ_D + seg * 32, 0, 1);
    }
    __syncthreads();

    // ---- 16 WMMAs: acc(16x16) = xtile(16x512,bf16) . etile^T
    v8f acc = {0.f,0.f,0.f,0.f,0.f,0.f,0.f,0.f};
    const unsigned* brow = &sB[m * LDS_STRIDE + (kbaseB >> 1)];
#pragma unroll
    for (int kc = 0; kc < VQ_KCHUNKS; ++kc) {
      union BU { uint4 q[2]; v16bf v; } bf;
      const uint4* bp = (const uint4*)(brow + kc * 16);
      bf.q[0] = bp[0];
      bf.q[1] = bp[2];
      acc = __builtin_amdgcn_wmma_f32_16x16x32_bf16(
          false, afrag[kc].v, false, bf.v, (short)0, acc, false, false);
    }

    // ---- Fused score + running argmax (per-lane: 8 rows, one code column).
    const float e2n  = sE2[m];
    const int   code = tile * 16 + m;
#pragma unroll
    for (int j = 0; j < 8; ++j) {
      float sc = 2.0f * acc[j] - e2n;
      if (sc > best[j] || (sc == best[j] && code < bidx[j])) {
        best[j] = sc; bidx[j] = code;
      }
    }
    __syncthreads();
  }

  // ---- Cross-lane argmax within each 16-lane half (C layout: rows 0-7 in
  // lanes 0-15, rows 8-15 in lanes 16-31).
#pragma unroll
  for (int j = 0; j < 8; ++j) {
#pragma unroll
    for (int mask = 1; mask <= 8; mask <<= 1) {
      float ob = __shfl_xor(best[j], mask, 32);
      int   oi = __shfl_xor(bidx[j], mask, 32);
      if (ob > best[j] || (ob == best[j] && oi < bidx[j])) {
        best[j] = ob; bidx[j] = oi;
      }
    }
  }
  if (lane == 0) {
#pragma unroll
    for (int j = 0; j < 8; ++j) sIdx[wave * 16 + j] = bidx[j];
  }
  if (lane == 16) {
#pragma unroll
    for (int j = 0; j < 8; ++j) sIdx[wave * 16 + 8 + j] = bidx[j];
  }
  __syncthreads();

  // ---- Histogram (fp32 counts are exact up to 2^24).
  if (lane < 16) atomicAdd(&hist[sIdx[wave * 16 + lane]], 1.0f);

  // ---- Gather fp32 codewords, write output, accumulate commit SSE.
  float ssq = 0.0f;
#pragma unroll 1
  for (int rr = 0; rr < 16; ++rr) {
    const int c = sIdx[wave * 16 + rr];
    const float* eline = embed + (size_t)c * VQ_D + lane * 16;
    const float* xline = x + (size_t)(rowBase + rr) * VQ_D + lane * 16;
    float*       oline = outq + (size_t)(rowBase + rr) * VQ_D + lane * 16;
#pragma unroll
    for (int q = 0; q < 4; ++q) {
      float4 e4 = ((const float4*)eline)[q];
      float4 x4 = ((const float4*)xline)[q];
      float dx = e4.x - x4.x, dy = e4.y - x4.y;
      float dz = e4.z - x4.z, dw = e4.w - x4.w;
      ssq += dx*dx + dy*dy + dz*dz + dw*dw;
      ((float4*)oline)[q] = e4;
    }
  }
#pragma unroll
  for (int mask = 1; mask <= 16; mask <<= 1)
    ssq += __shfl_xor(ssq, mask, 32);
  if (lane == 0) atomicAdd(lossSlot, ssq);
}

__global__ void vq_finalize_kernel(const float* __restrict__ hist,
                                   float* lossSlot, float* perpSlot) {
  __shared__ float red[256];
  const int tid = threadIdx.x;
  float s = 0.0f;
  for (int c = tid; c < VQ_K; c += 256) {
    float p = hist[c] * (1.0f / (float)VQ_N);
    s += p * __logf(p + 1e-10f);
  }
  red[tid] = s;
  __syncthreads();
  for (int off = 128; off > 0; off >>= 1) {
    if (tid < off) red[tid] += red[tid + off];
    __syncthreads();
  }
  if (tid == 0) {
    *perpSlot = __expf(-red[0]);
    *lossSlot = *lossSlot * (1.0f / (float)(VQ_N * VQ_D));  // mean over N*d
  }
}

extern "C" void kernel_launch(void* const* d_in, const int* in_sizes, int n_in,
                              void* d_out, int out_size, void* d_ws, size_t ws_size,
                              hipStream_t stream) {
  (void)in_sizes; (void)n_in; (void)out_size; (void)d_ws; (void)ws_size;
  const float* x     = (const float*)d_in[0];   // [8,4096,512] fp32
  const float* embed = (const float*)d_in[1];   // [2048,512]   fp32

  float* out      = (float*)d_out;
  float* outq     = out;                              // 16777216 floats
  float* hist     = out + (size_t)VQ_N * VQ_D;        // 2048 floats
  float* lossSlot = hist + VQ_K;                      // 1 float (raw SSE)
  float* perpSlot = lossSlot + 1;                     // 1 float

  vq_init_kernel<<<(VQ_K + 1 + 255) / 256, 256, 0, stream>>>(hist);
  vq_main_kernel<<<VQ_N / 128, 256, 0, stream>>>(x, embed, outq, hist, lossSlot);
  vq_finalize_kernel<<<1, 256, 0, stream>>>(hist, lossSlot, perpSlot);
}
